// GraphUpdateBlock_87084756894239
// MI455X (gfx1250) — compile-verified, matrix operand order checked by reading, not run on hardware
//
#include <hip/hip_runtime.h>
#include <hip/hip_bf16.h>

typedef __attribute__((ext_vector_type(16))) __bf16 v16bf;
typedef __attribute__((ext_vector_type(8)))  float  v8f;

namespace {
constexpr int Bn  = 8;
constexpr int Nn  = 4096;
constexpr int NT  = Bn * Nn;      // 32768 nodes
constexpr int EPn = 32768;
constexpr int En  = Bn * EPn;     // 262144 edges
constexpr int KP  = 288;          // 260 padded up to multiple of 32
}

__device__ __forceinline__ float sigf(float x) { return 1.f / (1.f + __expf(-x)); }
__device__ __forceinline__ float siluf(float x) { return x * sigf(x); }

__device__ __forceinline__ float waveSum(float v) {
  #pragma unroll
  for (int o = 16; o > 0; o >>= 1) v += __shfl_xor(v, o, 32);
  return v;
}

__device__ __forceinline__ void atomicMaxF(float* addr, float val) {
  if (val >= 0.f) atomicMax((int*)addr, __float_as_int(val));
  else            atomicMin((unsigned int*)addr, __float_as_uint(val));
}

// ---------------- packing kernels ----------------

__global__ void k_pack_x(const float* __restrict__ nf, const float* __restrict__ r,
                         const float* __restrict__ tw, const float* __restrict__ vp,
                         __bf16* __restrict__ xb) {
  int i = blockIdx.x * blockDim.x + threadIdx.x;
  if (i >= NT * KP) return;
  int nt = i / KP, k = i - nt * KP;
  float v = 0.f;
  if (k < 256)      v = nf[(size_t)nt * 256 + k];
  else if (k == 256) v = r[nt * 2 + 0];
  else if (k == 257) v = r[nt * 2 + 1];
  else if (k == 258) v = tw[nt];
  else if (k == 259) v = vp[nt];
  xb[(size_t)nt * KP + k] = (__bf16)v;
}

__global__ void k_f32_to_bf(const float* __restrict__ x, __bf16* __restrict__ o, int n) {
  int i = blockIdx.x * blockDim.x + threadIdx.x;
  if (i < n) o[i] = (__bf16)x[i];
}

// W [K, N] f32 -> Wt [N, KPcols] bf16 (transposed, zero padded in K)
__global__ void k_packwT(const float* __restrict__ W, __bf16* __restrict__ Wt,
                         int K, int N, int KPcols) {
  int i = blockIdx.x * blockDim.x + threadIdx.x;
  if (i >= N * KPcols) return;
  int n = i / KPcols, k = i - n * KPcols;
  float v = (k < K) ? W[(size_t)k * N + n] : 0.f;
  Wt[(size_t)n * KPcols + k] = (__bf16)v;
}

__global__ void k_cat3(const float* __restrict__ a, int na,
                       const float* __restrict__ b, int nb,
                       const float* __restrict__ c, int nc,
                       float* __restrict__ o) {
  int i = blockIdx.x * blockDim.x + threadIdx.x;
  if (i < na) o[i] = a[i];
  else if (i < na + nb) o[i] = b[i - na];
  else if (i < na + nb + nc) o[i] = c[i - na - nb];
}

// ---------------- WMMA GEMM with async-LDS double-buffered staging ----------------
// C[M,Ncols] f32 = A[M,K](bf16 row-major) * Bt[Ncols,K](bf16) + bias ; ACT: 0=none 1=silu
// block = 8 waves, macro-tile 128M x 64N. Per K-step(32): stage A(8 frags) + B(4 frags)
// of 1KB each into LDS via global_load_async_to_lds_b128 (3 per wave, pointers
// precomputed and bumped +64B/step), consume via ds_load_b128 (all 10 issued before
// the 4 back-to-back v_wmma_f32_16x16x32_bf16).

#define STAGE_BYTES 12288   // 12 frags * 1KB

__device__ __forceinline__ void wait_async0() {
  asm volatile("s_wait_asynccnt 0x0" ::: "memory");
}

__device__ __forceinline__ void async_cp16(uint32_t lds_off, const __bf16* g) {
  asm volatile("global_load_async_to_lds_b128 %0, %1, off"
               :: "v"(lds_off), "v"(g) : "memory");
}

template <int ACT>
__global__ __launch_bounds__(256)
void k_wmma_gemm(const __bf16* __restrict__ A, const __bf16* __restrict__ Bt,
                 const float* __restrict__ bias, float* __restrict__ C,
                 int M, int Ncols, int K) {
  __shared__ __align__(128) char smem[2 * STAGE_BYTES];
  const int lane = threadIdx.x;                  // 0..31
  const int wave = threadIdx.y;                  // 0..7 -> m sub-tile
  const int nBlocksN = Ncols >> 6;
  const int bM = blockIdx.x / nBlocksN, bN = blockIdx.x - bM * nBlocksN;
  const int m0 = bM << 7, n0 = bN << 6;
  const int half = lane >> 4, l15 = lane & 15;
  const uint32_t smemBase = (uint32_t)(uintptr_t)smem;   // LDS byte address (low 32 bits)

  // precompute this thread's 3 staging copy slots (advance +32 elems per K-step)
  const __bf16* gptr[3];
  uint32_t lbase[3];
  #pragma unroll
  for (int j = 0; j < 3; j++) {
    int t = (wave * 3 + j) * 32 + lane;          // 0..767
    int f = t >> 6;                              // frag 0..11  (0-7 = A, 8-11 = B)
    int q = t & 63;
    int l = q >> 1, c = q & 1;
    int row16 = l & 15, hf = l >> 4;
    lbase[j] = smemBase + (uint32_t)(f * 1024 + l * 32 + c * 16);
    gptr[j] = (f < 8 ? A  + (size_t)(m0 + f * 16 + row16) * K
                     : Bt + (size_t)(n0 + (f - 8) * 16 + row16) * K)
              + hf * 8 + c * 16;
  }
  auto stage = [&](int buf) {
    #pragma unroll
    for (int j = 0; j < 3; j++) {
      async_cp16(lbase[j] + (uint32_t)(buf * STAGE_BYTES), gptr[j]);
      gptr[j] += 32;
    }
  };

  v8f acc[4];
  #pragma unroll
  for (int s = 0; s < 4; s++)
    #pragma unroll
    for (int r = 0; r < 8; r++) acc[s][r] = 0.f;

  stage(0);

  union Frag16 { v16bf v; float4 q[2]; };
  const int nK = K >> 5;
  for (int kk = 0; kk < nK; kk++) {
    wait_async0();           // previous stage's copies (only outstanding ones) done
    __syncthreads();         // all waves staged cur buf; all reads of other buf done
    const char* cur = smem + (kk & 1) * STAGE_BYTES;
    if (kk + 1 < nK) stage((kk + 1) & 1);
    // issue all 10 ds_load_b128 first, then 4 back-to-back WMMAs
    Frag16 a, b[4];
    const float4* af = (const float4*)(cur + wave * 1024 + lane * 32);
    a.q[0] = af[0]; a.q[1] = af[1];
    #pragma unroll
    for (int s = 0; s < 4; s++) {
      const float4* bf = (const float4*)(cur + (8 + s) * 1024 + lane * 32);
      b[s].q[0] = bf[0]; b[s].q[1] = bf[1];
    }
    #pragma unroll
    for (int s = 0; s < 4; s++)
      acc[s] = __builtin_amdgcn_wmma_f32_16x16x32_bf16(
          false, a.v, false, b[s].v, (short)0, acc[s], false, false);
  }

  const int mw = m0 + wave * 16;
  #pragma unroll
  for (int s = 0; s < 4; s++) {
    int n = n0 + s * 16 + l15;
    float bv = bias ? bias[n] : 0.f;
    #pragma unroll
    for (int r = 0; r < 8; r++) {
      int m = mw + half * 8 + r;
      float v = acc[s][r] + bv;
      if (ACT == 1) v = siluf(v);
      C[(size_t)m * Ncols + n] = v;
    }
  }
}

// ---------------- attention ----------------

__global__ void k_init(float* agg, float* maxb, float* den, float* apacc) {
  int i = blockIdx.x * blockDim.x + threadIdx.x;
  if (i < NT * 256) agg[i] = 0.f;
  if (i < NT * 4) { maxb[i] = -1e30f; den[i] = 0.f; }
  if (i < 8) apacc[i] = 0.f;
}

__device__ __forceinline__ void edge_ids(const int* __restrict__ edges, int e,
                                         int& src, int& dst) {
  int b = e >> 15, j = e & (EPn - 1);
  src = edges[(size_t)b * 2 * EPn + j] + b * Nn;
  dst = edges[(size_t)b * 2 * EPn + EPn + j] + b * Nn;
}

__global__ void k_edge_logit(const int* __restrict__ edges, const float* __restrict__ ea,
                             const float* __restrict__ y1, const float* __restrict__ We,
                             const float* __restrict__ att,
                             float* __restrict__ logit, float* __restrict__ maxb) {
  int t = blockIdx.x * blockDim.x + threadIdx.x;
  if (t >= En * 4) return;
  int e = t >> 2, h = t & 3;
  int src, dst; edge_ids(edges, e, src, dst);
  float e0 = ea[e * 3], e1 = ea[e * 3 + 1], e2 = ea[e * 3 + 2];
  const float* xls = y1 + (size_t)src * 768 + 256 + h * 64;
  const float* xrd = y1 + (size_t)dst * 768 + 512 + h * 64;
  const float* w0 = We + h * 64, *w1 = We + 256 + h * 64, *w2 = We + 512 + h * 64;
  const float* ah = att + h * 64;
  float lg = 0.f;
  for (int c = 0; c < 64; c++) {
    float v = xls[c] + xrd[c] + e0 * w0[c] + e1 * w1[c] + e2 * w2[c];
    v = v > 0.f ? v : 0.2f * v;
    lg += v * ah[c];
  }
  logit[t] = lg;
  atomicMaxF(&maxb[dst * 4 + h], lg);
}

__global__ void k_edge_den(const int* __restrict__ edges, const float* __restrict__ logit,
                           const float* __restrict__ maxb, float* __restrict__ den) {
  int t = blockIdx.x * blockDim.x + threadIdx.x;
  if (t >= En * 4) return;
  int e = t >> 2, h = t & 3;
  int src, dst; edge_ids(edges, e, src, dst); (void)src;
  atomicAdd(&den[dst * 4 + h], __expf(logit[t] - maxb[dst * 4 + h]));
}

__global__ void k_edge_agg(const int* __restrict__ edges, const float* __restrict__ logit,
                           const float* __restrict__ maxb, const float* __restrict__ den,
                           const float* __restrict__ y1, float* __restrict__ agg) {
  int t = blockIdx.x * blockDim.x + threadIdx.x;
  if (t >= En * 4) return;
  int e = t >> 2, h = t & 3;
  int src, dst; edge_ids(edges, e, src, dst);
  float alpha = __expf(logit[t] - maxb[dst * 4 + h]) / (den[dst * 4 + h] + 1e-16f);
  const float* xls = y1 + (size_t)src * 768 + 256 + h * 64;
  float* ag = agg + (size_t)dst * 256 + h * 64;
  for (int c = 0; c < 64; c++) atomicAdd(&ag[c], xls[c] * alpha);
}

// ---------------- wave-per-node elementwise stages ----------------

__global__ __launch_bounds__(256)
void k_postgat(const float* __restrict__ agg, const float* __restrict__ y1,
               const float* __restrict__ gb, const float* __restrict__ g,
               const float* __restrict__ bt, __bf16* __restrict__ ob) {
  int lane = threadIdx.x;
  int node = blockIdx.x * blockDim.y + threadIdx.y;
  if (node >= NT) return;
  int ch0 = lane * 8;
  float v[8], s = 0.f, s2 = 0.f;
  #pragma unroll
  for (int t = 0; t < 8; t++) {
    v[t] = agg[(size_t)node * 256 + ch0 + t] + gb[ch0 + t];
    s += v[t]; s2 += v[t] * v[t];
  }
  s = waveSum(s); s2 = waveSum(s2);
  float m = s * (1.f / 256.f);
  float rs = rsqrtf(s2 * (1.f / 256.f) - m * m + 1e-5f);
  #pragma unroll
  for (int t = 0; t < 8; t++) {
    int ch = ch0 + t;
    float y = (v[t] - m) * rs * g[ch] + bt[ch];
    y = siluf(y) + y1[(size_t)node * 768 + ch];
    ob[(size_t)node * 256 + ch] = (__bf16)y;
  }
}

__global__ __launch_bounds__(256)
void k_ln_bf(const float* __restrict__ x, const float* __restrict__ g,
             const float* __restrict__ bt, __bf16* __restrict__ ob) {
  int lane = threadIdx.x;
  int node = blockIdx.x * blockDim.y + threadIdx.y;
  if (node >= NT) return;
  int ch0 = lane * 8;
  float v[8], s = 0.f, s2 = 0.f;
  #pragma unroll
  for (int t = 0; t < 8; t++) {
    v[t] = x[(size_t)node * 256 + ch0 + t];
    s += v[t]; s2 += v[t] * v[t];
  }
  s = waveSum(s); s2 = waveSum(s2);
  float m = s * (1.f / 256.f);
  float rs = rsqrtf(s2 * (1.f / 256.f) - m * m + 1e-5f);
  #pragma unroll
  for (int t = 0; t < 8; t++) {
    int ch = ch0 + t;
    ob[(size_t)node * 256 + ch] = (__bf16)((v[t] - m) * rs * g[ch] + bt[ch]);
  }
}

__global__ __launch_bounds__(256)
void k_gru(const float* __restrict__ gi, const float* __restrict__ gh,
           const float* __restrict__ hf, const float* __restrict__ g,
           const float* __restrict__ bt, float* __restrict__ ho,
           __bf16* __restrict__ hob) {
  int lane = threadIdx.x;
  int node = blockIdx.x * blockDim.y + threadIdx.y;
  if (node >= NT) return;
  int ch0 = lane * 8;
  const float* gir = gi + (size_t)node * 768;
  const float* ghr = gh + (size_t)node * 768;
  float v[8], s = 0.f, s2 = 0.f;
  #pragma unroll
  for (int t = 0; t < 8; t++) {
    int ch = ch0 + t;
    float rg = sigf(gir[ch] + ghr[ch]);
    float zg = sigf(gir[256 + ch] + ghr[256 + ch]);
    float ng = tanhf(gir[512 + ch] + rg * ghr[512 + ch]);
    float hv = (1.f - zg) * ng + zg * hf[(size_t)node * 256 + ch];
    v[t] = hv; s += hv; s2 += hv * hv;
  }
  s = waveSum(s); s2 = waveSum(s2);
  float m = s * (1.f / 256.f);
  float rs = rsqrtf(s2 * (1.f / 256.f) - m * m + 1e-5f);
  #pragma unroll
  for (int t = 0; t < 8; t++) {
    int ch = ch0 + t;
    float y = (v[t] - m) * rs * g[ch] + bt[ch];
    ho[(size_t)node * 256 + ch] = y;
    hob[(size_t)node * 256 + ch] = (__bf16)y;
  }
}

__global__ void k_heads(const float* __restrict__ hh,
                        const float* __restrict__ wh2, const float* __restrict__ wh2b,
                        const float* __restrict__ ap2, const float* __restrict__ ap2b,
                        const float* __restrict__ ad2, const float* __restrict__ ad2b,
                        float* __restrict__ conf, float* __restrict__ ad,
                        float* __restrict__ apacc) {
  int nt = blockIdx.x * blockDim.x + threadIdx.x;
  if (nt >= NT) return;
  const float* row = hh + (size_t)nt * 256;
  float s = wh2b[0];
  for (int j = 0; j < 128; j++) s += row[j] * wh2[j * 2];
  conf[nt] = sigf(s);
  s = ap2b[0];
  for (int j = 0; j < 64; j++) s += row[128 + j] * ap2[j];
  atomicAdd(&apacc[nt >> 12], sigf(s));
  s = ad2b[0];
  for (int j = 0; j < 64; j++) s += row[192 + j] * ad2[j];
  ad[nt] = sigf(s) * 0.1f + 1e-4f;
}

__global__ void k_fin_ap(const float* __restrict__ apacc, float* __restrict__ ap) {
  int i = threadIdx.x;
  if (i < Bn) ap[i] = apacc[i] * (1.f / 4096.f) * 0.1f + 1e-4f;
}

// ---------------- params (jax tree-flatten = alphabetical dict keys) ----------------
enum {
  P_ad1, P_ad1b, P_ad2, P_ad2b, P_ap1, P_ap1b, P_ap2, P_ap2b, P_att,
  P_b_hh, P_b_ih, P_gat_bias, P_lin_edge_w, P_lin_l_b, P_lin_l_w,
  P_lin_r_b, P_lin_r_w, P_norm_b, P_norm_g, P_norm_gat_b, P_norm_gat_g,
  P_norm_h_b, P_norm_h_g, P_proj_b, P_proj_w, P_res_b, P_res_w,
  P_w_hh, P_w_ih, P_wh1, P_wh1b, P_wh2, P_wh2b, P_COUNT
};
static const size_t g_psz[P_COUNT] = {
  16384, 64, 64, 1, 16384, 64, 64, 1, 256,
  768, 768, 256, 768, 256, 66560,
  256, 66560, 256, 256, 256, 256,
  256, 256, 256, 65536, 256, 66560,
  196608, 196608, 32768, 128, 256, 2
};

extern "C" void kernel_launch(void* const* d_in, const int* in_sizes, int n_in,
                              void* d_out, int out_size, void* d_ws, size_t ws_size,
                              hipStream_t stream) {
  const float* h_in  = (const float*)d_in[0];
  const float* nf    = (const float*)d_in[1];
  const float* r_in  = (const float*)d_in[2];
  const float* tw    = (const float*)d_in[3];
  const float* vp    = (const float*)d_in[4];
  const int*   edges = (const int*)d_in[5];
  const float* ea    = (const float*)d_in[6];

  const float* P[P_COUNT];
  if (n_in >= 7 + P_COUNT) {
    for (int i = 0; i < P_COUNT; i++) P[i] = (const float*)d_in[7 + i];
  } else {
    const float* base = (const float*)d_in[7];
    size_t off = 0;
    for (int i = 0; i < P_COUNT; i++) { P[i] = base + off; off += g_psz[i]; }
  }

  // workspace bump allocator (256B aligned)
  char* w = (char*)d_ws;
  auto alloc = [&](size_t bytes) { char* p = w; w += (bytes + 255) & ~(size_t)255; return p; };
  __bf16* x_bf    = (__bf16*)alloc((size_t)NT * KP * 2);
  __bf16* Wt1     = (__bf16*)alloc((size_t)768 * KP * 2);
  float*  bias1   = (float*) alloc(768 * 4);
  __bf16* WtProj  = (__bf16*)alloc((size_t)256 * 256 * 2);
  __bf16* WtIH    = (__bf16*)alloc((size_t)768 * 256 * 2);
  __bf16* WtHH    = (__bf16*)alloc((size_t)768 * 256 * 2);
  __bf16* WtHeads = (__bf16*)alloc((size_t)256 * 256 * 2);
  float*  biasH   = (float*) alloc(256 * 4);
  float*  bigA    = (float*) alloc((size_t)NT * 768 * 4);  // y1 -> gi -> headsH
  float*  bigB    = (float*) alloc((size_t)NT * 768 * 4);  // agg|proj_out -> gh
  float*  logit   = (float*) alloc((size_t)En * 4 * 4);
  float*  maxb    = (float*) alloc((size_t)NT * 4 * 4);
  float*  den     = (float*) alloc((size_t)NT * 4 * 4);
  __bf16* bf_tmp  = (__bf16*)alloc((size_t)NT * 256 * 2);  // out_bf -> h_new_bf
  __bf16* xsp_bf  = (__bf16*)alloc((size_t)NT * 256 * 2);
  __bf16* h_bf    = (__bf16*)alloc((size_t)NT * 256 * 2);
  float*  apacc   = (float*) alloc(32);

  float* agg      = bigB;
  float* proj_out = bigB + (size_t)NT * 256;
  float* out    = (float*)d_out;
  float* o_h    = out;
  float* o_conf = out + (size_t)NT * 256;
  float* o_ap   = o_conf + NT;
  float* o_ad   = o_ap + Bn;

  const int T = 256;
  auto cdiv = [](long a, long b) { return (unsigned)((a + b - 1) / b); };

  k_pack_x<<<cdiv((long)NT * KP, T), T, 0, stream>>>(nf, r_in, tw, vp, x_bf);
  k_f32_to_bf<<<cdiv((long)NT * 256, T), T, 0, stream>>>(h_in, h_bf, NT * 256);
  k_packwT<<<cdiv(256L * KP, T), T, 0, stream>>>(P[P_res_w],   Wt1,             260, 256, KP);
  k_packwT<<<cdiv(256L * KP, T), T, 0, stream>>>(P[P_lin_l_w], Wt1 + 256 * KP,  260, 256, KP);
  k_packwT<<<cdiv(256L * KP, T), T, 0, stream>>>(P[P_lin_r_w], Wt1 + 512 * KP,  260, 256, KP);
  k_packwT<<<cdiv(256L * 256, T), T, 0, stream>>>(P[P_proj_w], WtProj, 256, 256, 256);
  k_packwT<<<cdiv(768L * 256, T), T, 0, stream>>>(P[P_w_ih], WtIH, 256, 768, 256);
  k_packwT<<<cdiv(768L * 256, T), T, 0, stream>>>(P[P_w_hh], WtHH, 256, 768, 256);
  k_packwT<<<cdiv(128L * 256, T), T, 0, stream>>>(P[P_wh1], WtHeads,             256, 128, 256);
  k_packwT<<<cdiv(64L * 256, T), T, 0, stream>>>(P[P_ap1],  WtHeads + 128 * 256, 256, 64, 256);
  k_packwT<<<cdiv(64L * 256, T), T, 0, stream>>>(P[P_ad1],  WtHeads + 192 * 256, 256, 64, 256);
  k_cat3<<<3, T, 0, stream>>>(P[P_res_b], 256, P[P_lin_l_b], 256, P[P_lin_r_b], 256, bias1);
  k_cat3<<<1, T, 0, stream>>>(P[P_wh1b], 128, P[P_ap1b], 64, P[P_ad1b], 64, biasH);

  // y1 = [identity | xl | xr]     (async-LDS staged WMMA GEMM)
  k_wmma_gemm<0><<<(NT / 128) * (768 / 64), dim3(32, 8), 0, stream>>>(
      x_bf, Wt1, bias1, bigA, NT, 768, KP);

  // GATv2 attention (segment softmax over dst via atomics)
  k_init<<<cdiv((long)NT * 256, T), T, 0, stream>>>(agg, maxb, den, apacc);
  k_edge_logit<<<cdiv((long)En * 4, T), T, 0, stream>>>(edges, ea, bigA,
      P[P_lin_edge_w], P[P_att], logit, maxb);
  k_edge_den<<<cdiv((long)En * 4, T), T, 0, stream>>>(edges, logit, maxb, den);
  k_edge_agg<<<cdiv((long)En * 4, T), T, 0, stream>>>(edges, logit, maxb, den, bigA, agg);

  k_postgat<<<NT / 8, dim3(32, 8), 0, stream>>>(agg, bigA, P[P_gat_bias],
      P[P_norm_g], P[P_norm_b], bf_tmp);
  k_wmma_gemm<0><<<(NT / 128) * (256 / 64), dim3(32, 8), 0, stream>>>(
      bf_tmp, WtProj, P[P_proj_b], proj_out, NT, 256, 256);
  k_ln_bf<<<NT / 8, dim3(32, 8), 0, stream>>>(proj_out, P[P_norm_gat_g],
      P[P_norm_gat_b], xsp_bf);

  // GRU gates + fused cell/LN
  k_wmma_gemm<0><<<(NT / 128) * (768 / 64), dim3(32, 8), 0, stream>>>(
      xsp_bf, WtIH, P[P_b_ih], bigA, NT, 768, 256);
  k_wmma_gemm<0><<<(NT / 128) * (768 / 64), dim3(32, 8), 0, stream>>>(
      h_bf, WtHH, P[P_b_hh], bigB, NT, 768, 256);
  k_gru<<<NT / 8, dim3(32, 8), 0, stream>>>(bigA, bigB, h_in,
      P[P_norm_h_g], P[P_norm_h_b], o_h, bf_tmp);

  // heads: fused [wh1|ap1|ad1] GEMM with silu epilogue, then tiny layer-2 dots
  k_wmma_gemm<1><<<(NT / 128) * (256 / 64), dim3(32, 8), 0, stream>>>(
      bf_tmp, WtHeads, biasH, bigA, NT, 256, 256);
  k_heads<<<cdiv(NT, T), T, 0, stream>>>(bigA, P[P_wh2], P[P_wh2b],
      P[P_ap2], P[P_ap2b], P[P_ad2], P[P_ad2b], o_conf, o_ad, apacc);
  k_fin_ap<<<1, 32, 0, stream>>>(apacc, o_ap);

  (void)in_sizes; (void)out_size; (void)ws_size;
}
